// Edge_conv_58299886076589
// MI455X (gfx1250) — matrix operand order repaired and checked by reading, not compiled, not run on hardware
//
#include <hip/hip_runtime.h>
#include <hip/hip_bf16.h>

typedef __attribute__((ext_vector_type(16))) _Float16 v16h;
typedef __attribute__((ext_vector_type(8)))  float    v8f;

#define KNN   20
#define NPTS  4096
#define NBAT  8

// ==========================================================================
// Layout note: all intermediate activations are point-major  xpm[b][n][c]
// so channel runs are contiguous (vector loads everywhere). Only the
// original input (B,C,N) and the final output (B,CO,N) are channel-major.
// ==========================================================================

// ---- input transpose (B,C,N) -> point-major (B,N,C) ----------------------
__global__ void chw2nc_kernel(const float* __restrict__ x, float* __restrict__ xpm,
                              int C, int N) {
    int i = blockIdx.x * blockDim.x + threadIdx.x;   // b*N + n
    int b = i / N, n = i - b * N;
    for (int c = 0; c < C; ++c)
        xpm[((size_t)b * N + n) * C + c] = x[((size_t)b * C + c) * N + n];
}

// ---- squared norms -------------------------------------------------------
__global__ void norms_kernel(const float* __restrict__ xpm, float* __restrict__ xx,
                             int C, int N) {
    int i = blockIdx.x * blockDim.x + threadIdx.x;   // b*N + n
    const float* p = xpm + (size_t)i * C;
    float s = 0.f;
    for (int c = 0; c < C; ++c) { float v = p[c]; s += v * v; }
    xx[i] = s;
}

// ---- pre-swizzle WMMA B-fragments, one per (batch, 16-col tile, k-step) --
// B order used here: lane (half,l16): col=l16, K = half*16 + e (+32/step)
// Stored lane-major: 32 lanes x 16 f16 = 1 KB per (tile,step); each consumer
// lane reads its 32 contiguous bytes with two global_load_b128.
template<int C>
__global__ void packB_kernel(const float* __restrict__ xpm, _Float16* __restrict__ Bpack,
                             int N) {
    constexpr int KS = (C + 31) / 32;
    const int b = blockIdx.y, jt = blockIdx.x;
    const int t = threadIdx.x;            // 32*KS threads
    const int s = t >> 5, lane = t & 31;
    const int half = lane >> 4, l16 = lane & 15;
    const float* xc = xpm + ((size_t)b * N + jt * 16 + l16) * C;   // contiguous channels
    _Float16* out = Bpack + (((size_t)b * (N / 16) + jt) * KS + s) * 512 + lane * 16;
#pragma unroll
    for (int e = 0; e < 16; ++e) {
        int k = s * 32 + half * 16 + e;
        out[e] = (k < C) ? (_Float16)xc[k] : (_Float16)0.f;
    }
}

// ---- kNN: score s(i,j) = 2*<x_i,x_j> - ||x_j||^2 (row norm is rank-inert)
// One wave32 per 16-row tile; inner loop = vector loads + WMMA only.
template<int C>
__global__ void __launch_bounds__(32)
knn_kernel(const float* __restrict__ xpm, const float* __restrict__ xx,
           const _Float16* __restrict__ Bpack, int* __restrict__ idx, int N) {
    constexpr int KS = (C + 31) / 32;
    const int b       = blockIdx.y;
    const int rowBase = blockIdx.x * 16;
    const int lane    = threadIdx.x;
    const int half    = lane >> 4, l16 = lane & 15;
    const int ntiles  = N / 16;

    __shared__ float tile[16][17];
    __shared__ float topv[16][KNN];
    __shared__ int   topi[16][KNN];

    if (lane < 16)
        for (int k = 0; k < KNN; ++k) { topv[lane][k] = -3.0e38f; topi[lane][k] = 0; }

    // A fragment: row rowBase+l16; k order used here:
    //   elem e -> K = half*16+e (+32 per k-step), matching Bpack
    // point-major memory => one contiguous 16-float chunk per k-step.
    const float* xr = xpm + ((size_t)b * N + rowBase + l16) * C;
    v16h a[KS];
#pragma unroll
    for (int s = 0; s < KS; ++s) {
        v16h av;
#pragma unroll
        for (int e = 0; e < 16; ++e) {
            int k = s * 32 + half * 16 + e;
            av[e] = (k < C) ? (_Float16)xr[k] : (_Float16)0.f;
        }
        a[s] = av;
    }

    const float* xxb = xx + (size_t)b * N;
    const _Float16* bpB = Bpack + (size_t)b * ntiles * KS * 512 + lane * 16;
    __syncthreads();

    for (int jt = 0; jt < ntiles; ++jt) {
        const _Float16* bp = bpB + (size_t)jt * KS * 512;
        v8f acc = {};
#pragma unroll
        for (int s = 0; s < KS; ++s) {
            v16h bv = *(const v16h*)(bp + s * 512);
            acc = __builtin_amdgcn_wmma_f32_16x16x32_f16(
                false, a[s], false, bv, (short)0, acc, false, false);
        }
        if (jt + 1 < ntiles)
            __builtin_prefetch(bpB + (size_t)(jt + 1) * KS * 512, 0, 3);

        // C/D layout: VGPR i, lanes 0-15 -> M=i, lanes 16-31 -> M=i+8; N=l16
        float xxc = xxb[jt * 16 + l16];
#pragma unroll
        for (int i = 0; i < 8; ++i)
            tile[half * 8 + i][l16] = 2.f * acc[i] - xxc;
        __syncthreads();

        if (lane < 16) {
            float thr = topv[lane][KNN - 1];
            for (int c16 = 0; c16 < 16; ++c16) {
                float v = tile[lane][c16];
                if (v > thr) {
                    int p = KNN - 1;
                    while (p > 0 && topv[lane][p - 1] < v) {   // stable ties
                        topv[lane][p] = topv[lane][p - 1];
                        topi[lane][p] = topi[lane][p - 1];
                        --p;
                    }
                    topv[lane][p] = v;
                    topi[lane][p] = jt * 16 + c16;
                    thr = topv[lane][KNN - 1];
                }
            }
        }
        __syncthreads();
    }

    if (lane < 16) {
        int* op = idx + ((size_t)b * N + rowBase + lane) * KNN;
        for (int k = 0; k < KNN; ++k) op[k] = topi[lane][k];
    }
}

// ---- BN fold + weight split ---------------------------------------------
__global__ void prep_kernel(const float* __restrict__ W, const float* __restrict__ bb,
                            const float* __restrict__ gamma, const float* __restrict__ beta,
                            const float* __restrict__ mean, const float* __restrict__ var,
                            _Float16* __restrict__ G1, _Float16* __restrict__ G2,
                            float* __restrict__ cshift, int C, int CO) {
    int i = blockIdx.x * blockDim.x + threadIdx.x;
    if (i >= CO * C) return;
    int o = i / C, c = i - o * C;
    float s  = gamma[o] * rsqrtf(var[o] + 1e-5f);
    float w1 = W[(size_t)o * 2 * C + c];
    float w2 = W[(size_t)o * 2 * C + C + c];
    G1[i] = (_Float16)(s * w1);
    G2[i] = (_Float16)(s * (w2 - w1));
    if (c == 0) cshift[o] = s * bb[o] + beta[o] - mean[o] * s;
}

// ---- dense GEMM R[b][n][o] = sum_c G[o][c] * x[b][n][c] (reuses Bpack) ---
template<int C>
__global__ void __launch_bounds__(32)
gemm_kernel(const _Float16* __restrict__ G, const _Float16* __restrict__ Bpack,
            float* __restrict__ R, int CO, int N) {
    constexpr int KS = (C + 31) / 32;
    const int b = blockIdx.z, jt = blockIdx.x, ot = blockIdx.y;
    const int lane = threadIdx.x;
    const int half = lane >> 4, l16 = lane & 15;
    const _Float16* gr = G + (size_t)(ot * 16 + l16) * C;   // contiguous f16 row
    const _Float16* bp = Bpack + (((size_t)b * (N / 16) + jt) * KS) * 512 + lane * 16;

    v8f acc = {};
#pragma unroll
    for (int s = 0; s < KS; ++s) {
        v16h av;
#pragma unroll
        for (int e = 0; e < 16; ++e) {
            int k = s * 32 + half * 16 + e;
            av[e] = (k < C) ? gr[k] : (_Float16)0.f;
        }
        v16h bv = *(const v16h*)(bp + s * 512);
        acc = __builtin_amdgcn_wmma_f32_16x16x32_f16(
            false, av, false, bv, (short)0, acc, false, false);
    }
    // lane holds o = ot*16 + half*8 + i (contiguous), n = jt*16 + l16
    float* rp = R + (((size_t)b * N + jt * 16 + l16) * CO) + ot * 16 + half * 8;
#pragma unroll
    for (int i = 0; i < 8; ++i) rp[i] = acc[i];
}

// ---- fused gather + lrelu + max over K neighbors -------------------------
template<bool FINAL_CHW>
__global__ void fusemax_kernel(const float* __restrict__ P, const float* __restrict__ Q,
                               const float* __restrict__ cshift, const int* __restrict__ idx,
                               float* __restrict__ xout, int CO, int N) {
    const int b = blockIdx.y, n = blockIdx.x, o = threadIdx.x;   // CO threads
    const int* ip = idx + ((size_t)b * N + n) * KNN;
    float q = Q[((size_t)b * N + n) * CO + o] + cshift[o];
    float m = -3.0e38f;
#pragma unroll
    for (int k = 0; k < KNN; ++k) {
        int j = ip[k];
        float v = P[((size_t)b * N + j) * CO + o] + q;   // contiguous gather column
        v = (v > 0.f) ? v : 0.2f * v;
        m = fmaxf(m, v);
    }
    if (FINAL_CHW) xout[((size_t)b * CO + o) * N + n] = m;   // (B,CO,N) output
    else           xout[((size_t)b * N + n) * CO + o] = m;   // point-major
}

// ==========================================================================
extern "C" void kernel_launch(void* const* d_in, const int* in_sizes, int n_in,
                              void* d_out, int out_size, void* d_ws, size_t ws_size,
                              hipStream_t stream) {
    const int N = NPTS, B = NBAT;
    const int CO_ARR[3] = {64, 64, 128};
    const int COMAX = 128, CMAX = 64, NT = N / 16;

    char* ws = (char*)d_ws;
    size_t off = 0;
    auto carve = [&](size_t bytes) -> void* {
        void* p = ws + off;
        off = (off + bytes + 255) & ~(size_t)255;
        return p;
    };
    float*    xx     = (float*)   carve((size_t)B * N * sizeof(float));
    int*      idxbuf = (int*)     carve((size_t)B * N * KNN * sizeof(int));
    _Float16* Bpack  = (_Float16*)carve((size_t)B * NT * 2 * 512 * sizeof(_Float16));
    _Float16* G1     = (_Float16*)carve((size_t)COMAX * CMAX * sizeof(_Float16));
    _Float16* G2     = (_Float16*)carve((size_t)COMAX * CMAX * sizeof(_Float16));
    float*    cshift = (float*)   carve((size_t)COMAX * sizeof(float));
    float*    P      = (float*)   carve((size_t)B * N * COMAX * sizeof(float));
    float*    Q      = (float*)   carve((size_t)B * N * COMAX * sizeof(float));
    float*    xpmA   = (float*)   carve((size_t)B * N * CMAX * sizeof(float));
    float*    xpmB   = (float*)   carve((size_t)B * N * CMAX * sizeof(float));
    (void)ws_size; (void)in_sizes; (void)n_in; (void)out_size;

    // input (B,3,N) -> point-major
    chw2nc_kernel<<<(B * N) / 256, 256, 0, stream>>>((const float*)d_in[0], xpmA, 3, N);

    const float* xin = xpmA;
    int C = 3;
    float* pmOut[3] = {xpmB, xpmA, nullptr};

    for (int L = 0; L < 3; ++L) {
        const float* W     = (const float*)d_in[1 + 6 * L + 0];
        const float* bbias = (const float*)d_in[1 + 6 * L + 1];
        const float* gamma = (const float*)d_in[1 + 6 * L + 2];
        const float* beta  = (const float*)d_in[1 + 6 * L + 3];
        const float* mean  = (const float*)d_in[1 + 6 * L + 4];
        const float* var   = (const float*)d_in[1 + 6 * L + 5];
        const int CO = CO_ARR[L];

        norms_kernel<<<(B * N) / 256, 256, 0, stream>>>(xin, xx, C, N);

        if (C == 3) {
            packB_kernel<3><<<dim3(NT, B), 32, 0, stream>>>(xin, Bpack, N);
            knn_kernel<3><<<dim3(NT, B), 32, 0, stream>>>(xin, xx, Bpack, idxbuf, N);
        } else {
            packB_kernel<64><<<dim3(NT, B), 64, 0, stream>>>(xin, Bpack, N);
            knn_kernel<64><<<dim3(NT, B), 32, 0, stream>>>(xin, xx, Bpack, idxbuf, N);
        }

        int tot = CO * C;
        prep_kernel<<<(tot + 255) / 256, 256, 0, stream>>>(
            W, bbias, gamma, beta, mean, var, G1, G2, cshift, C, CO);

        if (C == 3) {
            gemm_kernel<3><<<dim3(NT, CO / 16, B), 32, 0, stream>>>(G1, Bpack, P, CO, N);
            gemm_kernel<3><<<dim3(NT, CO / 16, B), 32, 0, stream>>>(G2, Bpack, Q, CO, N);
        } else {
            gemm_kernel<64><<<dim3(NT, CO / 16, B), 32, 0, stream>>>(G1, Bpack, P, CO, N);
            gemm_kernel<64><<<dim3(NT, CO / 16, B), 32, 0, stream>>>(G2, Bpack, Q, CO, N);
        }

        if (L == 2) {
            fusemax_kernel<true><<<dim3(N, B), CO, 0, stream>>>(
                P, Q, cshift, idxbuf, (float*)d_out, CO, N);
        } else {
            fusemax_kernel<false><<<dim3(N, B), CO, 0, stream>>>(
                P, Q, cshift, idxbuf, pmOut[L], CO, N);
            xin = pmOut[L];
        }
        C = CO;
    }
}
